// MultiHeadAttention_83099027243652
// MI455X (gfx1250) — matrix operand order repaired and checked by reading, not compiled, not run on hardware
//
#include <hip/hip_runtime.h>
#include <cstdint>

#define DEV __device__ __forceinline__

typedef __attribute__((ext_vector_type(16))) __bf16        v16bf;
typedef __attribute__((ext_vector_type(8)))  float         v8f;
typedef __attribute__((ext_vector_type(8)))  unsigned int  v8u;
typedef __attribute__((ext_vector_type(4)))  unsigned int  u32x4;
typedef __attribute__((ext_vector_type(8)))  int           i32x8;
typedef __attribute__((ext_vector_type(4)))  int           i32x4;

static constexpr int B_ = 2, S_ = 2048, EMB = 512, HEADS_ = 8, HD = 64;
static constexpr int H3 = 3 * EMB;  // 1536
static constexpr float NEGF = -9.0e15f;

// ---------- helpers ----------
DEV unsigned short f2bf(float f) {
  unsigned u = __builtin_bit_cast(unsigned, f);
  unsigned r = u + 0x7fffu + ((u >> 16) & 1u);   // RNE f32 -> bf16
  return (unsigned short)(r >> 16);
}
DEV unsigned pk2(float lo, float hi) {
  return (unsigned)f2bf(lo) | ((unsigned)f2bf(hi) << 16);
}
DEV v16bf frag_from_lds16(const unsigned short* p) {   // 16 consecutive bf16
  v8u u;
  const unsigned* q = (const unsigned*)p;
#pragma unroll
  for (int i = 0; i < 8; ++i) u[i] = q[i];
  return __builtin_bit_cast(v16bf, u);
}
DEV v16bf frag_from_lds8x2(const unsigned short* p0, const unsigned short* p1) {
  v8u u;
  const unsigned* a = (const unsigned*)p0;
  const unsigned* b = (const unsigned*)p1;
#pragma unroll
  for (int i = 0; i < 4; ++i) { u[i] = a[i]; u[i + 4] = b[i]; }
  return __builtin_bit_cast(v16bf, u);
}
DEV v8f wmma_bf16(v16bf a, v16bf b, v8f c) {
  return __builtin_amdgcn_wmma_f32_16x16x32_bf16(false, a, false, b, (short)0, c,
                                                 false, false);
}

// ---------- Tensor Data Mover: 2D bf16 tile -> LDS, with LDS row padding ----------
// D# layout per CDNA5 ISA ch.8 (group0 128b, group1 256b; higher groups zero => <=2D).
// pad_interval code: 0=2,1=4,2=8,3=16,4=32... dwords; pad_amount code: n => n+1 dwords.
// This toolchain exposes the 6-arg builtin: (g0, g1, g2, g3, g4, cpol).
DEV void tdm_load_2d_bf16(unsigned lds_off_bytes, const unsigned short* gptr,
                          unsigned tensor_d0, unsigned tensor_d1,
                          unsigned stride_elems, unsigned tile_d0, unsigned tile_d1,
                          unsigned pad_interval_code, unsigned pad_amount_code) {
  unsigned long long ga = (unsigned long long)(uintptr_t)gptr;
  u32x4 g0;
  g0[0] = 1u;                                        // count=1, user mode
  g0[1] = lds_off_bytes;                             // lds_addr
  g0[2] = (unsigned)(ga & 0xffffffffu);              // global_addr[31:0]
  g0[3] = (unsigned)((ga >> 32) & 0x1ffffffu) | (2u << 30);  // addr[56:32] | type=2
  i32x8 g1;
  g1[0] = (int)((1u << 16)                           // data_size = 2 bytes
              | (1u << 20)                           // pad_enable
              | (pad_interval_code << 22)
              | (pad_amount_code << 25));
  g1[1] = (int)((tensor_d0 & 0xffffu) << 16);        // tensor_dim0[15:0] @ bit48
  g1[2] = (int)(((tensor_d0 >> 16) & 0xffffu) | ((tensor_d1 & 0xffffu) << 16));
  g1[3] = (int)(((tensor_d1 >> 16) & 0xffffu) | ((tile_d0 & 0xffffu) << 16));
  g1[4] = (int)(tile_d1 & 0xffffu);                  // tile_dim1; tile_dim2=0
  g1[5] = (int)stride_elems;                         // tensor_dim0_stride lo32
  g1[6] = 0;
  g1[7] = 0;
  i32x4 z4 = {0, 0, 0, 0};
  i32x8 z8 = {0, 0, 0, 0, 0, 0, 0, 0};
  __builtin_amdgcn_tensor_load_to_lds(g0, g1, z4, z4, z8, 0);
}

// ---------- tiled GEMM: C[M,N] = A[M,K] * W[K,N] + bias ----------
// AT = float (convert to bf16 in staging) or ushort (bf16 bits; staged via TDM).
// CT = float or ushort(bf16). block = 256 thr (8 waves, 4x2), tile 128x128, Kstep 32.
template <typename AT, typename CT>
__global__ __launch_bounds__(256) void gemm_wmma(const AT* __restrict__ A,
                                                 const float* __restrict__ W,
                                                 const float* __restrict__ bias,
                                                 CT* __restrict__ C, int K, int N) {
  constexpr int AP = 36;  // A tile pitch (ushorts): 32 data + 4 pad (=16+2 dwords)
  constexpr int BP = 34;  // W tile pitch, stored transposed [col][k]
  __shared__ unsigned short sA[128 * AP];
  __shared__ unsigned short sB[128 * BP];

  const int tid = threadIdx.x, lane = tid & 31, wave = tid >> 5;
  const int hl = lane >> 4, ln = lane & 15;
  const int wm = wave >> 1, wn = wave & 1;
  const int row0 = blockIdx.y * 128, col0 = blockIdx.x * 128;

  v8f acc[2][4] = {};

  for (int k0 = 0; k0 < K; k0 += 32) {
    if constexpr (__is_same(AT, unsigned short)) {
      // Async DMA: 128x32 bf16 tile, row stride K, LDS pitch 36 ushorts.
      if (wave == 0)
        tdm_load_2d_bf16((unsigned)(uintptr_t)&sA[0],
                         A + (size_t)row0 * K + k0,
                         /*tensor_d0=*/32, /*tensor_d1=*/128, /*stride=*/(unsigned)K,
                         /*tile_d0=*/32, /*tile_d1=*/128,
                         /*pad_interval(16dw)=*/3, /*pad_amount(2dw)=*/1);
    } else {
      // f32 path: convert while staging. thread t: row t/2, 16 cols.
      int r = tid >> 1, cs = (tid & 1) * 16;
      const float* src = A + (size_t)(row0 + r) * K + k0 + cs;
      float4 f0 = *(const float4*)(src + 0);
      float4 f1 = *(const float4*)(src + 4);
      float4 f2 = *(const float4*)(src + 8);
      float4 f3 = *(const float4*)(src + 12);
      unsigned* d = (unsigned*)&sA[r * AP + cs];
      d[0] = pk2(f0.x, f0.y); d[1] = pk2(f0.z, f0.w);
      d[2] = pk2(f1.x, f1.y); d[3] = pk2(f1.z, f1.w);
      d[4] = pk2(f2.x, f2.y); d[5] = pk2(f2.z, f2.w);
      d[6] = pk2(f3.x, f3.y); d[7] = pk2(f3.z, f3.w);
      if (k0 + 32 < K)
        __builtin_prefetch(src + 32, 0, 1);  // global_prefetch_b8 next tile
    }
    // stage W transposed (sB[col][k]) so B fragments are contiguous ds reads
    {
      int kr = tid >> 3, cs = (tid & 7) * 16;
      const float* src = W + (size_t)(k0 + kr) * N + col0 + cs;
#pragma unroll
      for (int j = 0; j < 16; j += 4) {
        float4 f = *(const float4*)(src + j);
        sB[(cs + j + 0) * BP + kr] = f2bf(f.x);
        sB[(cs + j + 1) * BP + kr] = f2bf(f.y);
        sB[(cs + j + 2) * BP + kr] = f2bf(f.z);
        sB[(cs + j + 3) * BP + kr] = f2bf(f.w);
      }
    }
    if constexpr (__is_same(AT, unsigned short)) {
      if (wave == 0) __builtin_amdgcn_s_wait_tensorcnt((short)0);
    }
    __syncthreads();

    v16bf af[2], bfr[4];
#pragma unroll
    for (int i = 0; i < 2; ++i) {
      const unsigned short* p = &sA[(wm * 32 + i * 16 + ln) * AP + 8 * hl];
      af[i] = frag_from_lds8x2(p, p + 16);
    }
#pragma unroll
    for (int j = 0; j < 4; ++j)
      bfr[j] = frag_from_lds8x2(&sB[(wn * 64 + j * 16 + ln) * BP + 8 * hl], &sB[(wn * 64 + j * 16 + ln) * BP + 8 * hl + 16]);
#pragma unroll
    for (int i = 0; i < 2; ++i)
#pragma unroll
      for (int j = 0; j < 4; ++j) acc[i][j] = wmma_bf16(af[i], bfr[j], acc[i][j]);
    __syncthreads();
  }

  // epilogue: C/D layout element (M = r + 8*hl, N = ln)
#pragma unroll
  for (int j = 0; j < 4; ++j) {
    int col = col0 + wn * 64 + j * 16 + ln;
    float bb = bias[col];
#pragma unroll
    for (int i = 0; i < 2; ++i) {
      int rowb = row0 + wm * 32 + i * 16 + 8 * hl;
#pragma unroll
      for (int r = 0; r < 8; ++r) {
        float v = acc[i][j][r] + bb;
        if constexpr (__is_same(CT, unsigned short))
          C[(size_t)(rowb + r) * N + col] = f2bf(v);
        else
          C[(size_t)(rowb + r) * N + col] = v;
      }
    }
  }
}

// ---------- banded attention (qkv and ctx are bf16 bits) ----------
// block = 128 thr (4 waves); block owns (b, h, 64 queries); wave owns 16 queries.
__global__ __launch_bounds__(128) void band_attn(const unsigned short* __restrict__ qkv,
                                                 const int* __restrict__ pmask,
                                                 unsigned short* __restrict__ ctx) {
  constexpr int KP = 66;    // sK pitch [key][d]: 64 data + 2 pad (=32+1 dwords)
  constexpr int VP = 146;   // sVt pitch [d][key]
  constexpr int PP = 96;    // sP pitch [m][c]: 80 probs + 16 zero pad
  __shared__ unsigned short sK[128 * KP];
  __shared__ unsigned short sVt[HD * VP];
  __shared__ unsigned short sP[4][16 * PP];

  const int b = blockIdx.z, h = blockIdx.y;
  const int qbase = blockIdx.x * 64;
  const int tid = threadIdx.x, wave = tid >> 5, lane = tid & 31;
  const int hl = lane >> 4, ln = lane & 15;
  const int kb = qbase - 32;
  const unsigned short* base = qkv + (size_t)b * S_ * H3;

  // ---- K band via Tensor Data Mover: tile (rows x 64) bf16, stride 1536 ----
  const int head = (kb < 0) ? -kb : 0;       // negative keys (first block only)
  const int row0 = kb + head;
  if (head > 0) {                             // pre-zero LDS rows for keys < 0
    for (int i = tid; i < head * KP; i += 128) sK[i] = 0;
  }
  if (wave == 0) {
    tdm_load_2d_bf16((unsigned)(uintptr_t)&sK[(size_t)head * KP],
                     base + (size_t)row0 * H3 + h * (3 * HD) + HD,
                     /*tensor_d0=*/HD, /*tensor_d1=*/(unsigned)(S_ - row0),  // tail OOB -> 0
                     /*stride=*/H3, /*tile_d0=*/HD, /*tile_d1=*/(unsigned)(128 - head),
                     /*pad_interval(32dw)=*/4, /*pad_amount(1dw)=*/0);
  }
  // ---- V band staged transposed by hand (bf16 copy, no conversion) ----
  {
    int key = kb + tid;
    bool inr = (key >= 0) && (key < S_);
    const unsigned short* vp =
        base + (size_t)(inr ? key : 0) * H3 + h * (3 * HD) + 2 * HD;
#pragma unroll
    for (int c = 0; c < HD; c += 8) {
      uint4 vv = inr ? *(const uint4*)(vp + c) : make_uint4(0, 0, 0, 0);
      unsigned vals[4] = {vv.x, vv.y, vv.z, vv.w};
#pragma unroll
      for (int j = 0; j < 4; ++j) {
        sVt[(c + 2 * j + 0) * VP + tid] = (unsigned short)(vals[j] & 0xffffu);
        sVt[(c + 2 * j + 1) * VP + tid] = (unsigned short)(vals[j] >> 16);
      }
    }
    if (tid < 16) {  // pad key slots 128..143 (hit only by zero P columns)
#pragma unroll
      for (int d0 = 0; d0 < HD; ++d0) sVt[d0 * VP + 128 + tid] = 0;
    }
  }
  if (wave == 0) __builtin_amdgcn_s_wait_tensorcnt((short)0);
  __syncthreads();

  const int q0 = qbase + wave * 16;

  // Q fragments: raw bf16 bits straight from global (A-matrix layout)
  v16bf qf[2];
  {
    const unsigned short* qrow = base + (size_t)(q0 + ln) * H3 + h * (3 * HD);
#pragma unroll
    for (int kk = 0; kk < 2; ++kk) {
      int d0 = kk * 32 + 8 * hl;
      uint4 a = *(const uint4*)(qrow + d0);
      uint4 c = *(const uint4*)(qrow + d0 + 16);
      v8u u;
      u[0] = a.x; u[1] = a.y; u[2] = a.z; u[3] = a.w;
      u[4] = c.x; u[5] = c.y; u[6] = c.z; u[7] = c.w;
      qf[kk] = __builtin_bit_cast(v16bf, u);
    }
  }

  // scores = Q * K^T over 5 key tiles (keys q0-32 .. q0+47)
  float sc[5][8];
#pragma unroll
  for (int kt = 0; kt < 5; ++kt) {
    v8f a = {};
    int krow = wave * 16 + kt * 16 + ln;
#pragma unroll
    for (int kk = 0; kk < 2; ++kk)
      a = wmma_bf16(qf[kk], frag_from_lds8x2(&sK[krow * KP + kk * 32 + 8 * hl], &sK[krow * KP + kk * 32 + 8 * hl + 16]), a);
    int key = kb + krow;
    bool kv = (key >= 0) && (key < S_) && (pmask[b * S_ + key] != 0);
#pragma unroll
    for (int r = 0; r < 8; ++r) {
      int q = q0 + r + 8 * hl;
      bool ok = kv && (key >= q - 32) && (key <= q + 32);
      sc[kt][r] = ok ? a[r] * 0.125f : NEGF;  // 1/sqrt(64)
    }
  }

  // safe softmax per row (row spans the 16 lanes of one half-wave group)
  float pv[5][8];
#pragma unroll
  for (int r = 0; r < 8; ++r) {
    float mx = sc[0][r];
#pragma unroll
    for (int kt = 1; kt < 5; ++kt) mx = fmaxf(mx, sc[kt][r]);
#pragma unroll
    for (int off = 1; off < 16; off <<= 1) mx = fmaxf(mx, __shfl_xor(mx, off, 32));
    float sum = 0.f;
#pragma unroll
    for (int kt = 0; kt < 5; ++kt) {
      float e = __expf(sc[kt][r] - mx);
      pv[kt][r] = e;
      sum += e;
    }
#pragma unroll
    for (int off = 1; off < 16; off <<= 1) sum += __shfl_xor(sum, off, 32);
    float inv = 1.0f / sum;
#pragma unroll
    for (int kt = 0; kt < 5; ++kt) pv[kt][r] *= inv;
  }

  // relayout P through LDS (C-layout -> A-layout), zero-pad cols 80..95
  {
    unsigned short* pw = &sP[wave][0];
#pragma unroll
    for (int r = 0; r < 8; ++r) {
      int m = r + 8 * hl;
#pragma unroll
      for (int kt = 0; kt < 5; ++kt) pw[m * PP + kt * 16 + ln] = f2bf(pv[kt][r]);
      pw[m * PP + 80 + ln] = 0;
    }
  }
  __syncthreads();

  // O = P(16x96) * Vband(96x64)
  v8f o[4] = {};
  const unsigned short* pw = &sP[wave][0];
#pragma unroll
  for (int kc = 0; kc < 3; ++kc) {
    const unsigned short* pa0 = &pw[ln * PP + kc * 32 + 8 * hl];
    v16bf pa = frag_from_lds8x2(pa0, pa0 + 16);
#pragma unroll
    for (int nt = 0; nt < 4; ++nt) {
      v16bf vb =
          frag_from_lds8x2(&sVt[(nt * 16 + ln) * VP + wave * 16 + kc * 32 + 8 * hl], &sVt[(nt * 16 + ln) * VP + wave * 16 + kc * 32 + 8 * hl + 16]);
      o[nt] = wmma_bf16(pa, vb, o[nt]);
    }
  }

  // epilogue: apply query padding mask, write context bf16 [B,S,EMB]
#pragma unroll
  for (int nt = 0; nt < 4; ++nt) {
    int d = nt * 16 + ln;
#pragma unroll
    for (int r = 0; r < 8; ++r) {
      int q = q0 + r + 8 * hl;
      float pmq = (pmask[b * S_ + q] != 0) ? 1.f : 0.f;
      ctx[((size_t)b * S_ + q) * EMB + h * HD + d] = f2bf(o[nt][r] * pmq);
    }
  }
}

// ---------- host launcher ----------
extern "C" void kernel_launch(void* const* d_in, const int* in_sizes, int n_in,
                              void* d_out, int out_size, void* d_ws, size_t ws_size,
                              hipStream_t stream) {
  (void)in_sizes; (void)n_in; (void)out_size; (void)ws_size;
  const float* x     = (const float*)d_in[0];
  const int*   pmask = (const int*)d_in[1];
  const float* Wqkv  = (const float*)d_in[2];
  const float* bqkv  = (const float*)d_in[3];
  const float* Wo    = (const float*)d_in[4];
  const float* bo    = (const float*)d_in[5];
  float* out = (float*)d_out;

  unsigned short* qkv = (unsigned short*)d_ws;           // [B*S,1536] bf16
  unsigned short* ctx = qkv + (size_t)B_ * S_ * H3;      // [B*S, 512] bf16

  // 1) qkv(bf16) = x @ Wqkv + bqkv   (4096 x 512 x 1536)
  gemm_wmma<float, unsigned short>
      <<<dim3(H3 / 128, (B_ * S_) / 128), 256, 0, stream>>>(x, Wqkv, bqkv, qkv,
                                                            /*K=*/512, /*N=*/H3);

  // 2) banded attention (TDM-staged K band) -> ctx(bf16)
  band_attn<<<dim3(S_ / 64, HEADS_, B_), 128, 0, stream>>>(qkv, pmask, ctx);

  // 3) out(f32) = ctx @ Wo + bo      (4096 x 512 x 512, TDM-staged A tiles)
  gemm_wmma<unsigned short, float>
      <<<dim3(EMB / 128, (B_ * S_) / 128), 256, 0, stream>>>(ctx, Wo, bo, out,
                                                             /*K=*/512, /*N=*/EMB);
}
